// RNN_chaotic_30004641530235
// MI455X (gfx1250) — matrix operand order repaired and checked by reading, not compiled, not run on hardware
//
#include <hip/hip_runtime.h>
#include <hip/hip_bf16.h>

typedef __attribute__((ext_vector_type(16))) __bf16 v16bf;
typedef __attribute__((ext_vector_type(8)))  __bf16 v8bf;
typedef __attribute__((ext_vector_type(8)))  float  v8f;

constexpr int kT = 512, kB = 64, kI = 256, kH = 2048, kO = 64;

// ---- workspace layout (bytes) ----
constexpr size_t OFF_INBF = 0;                                  // [T*B, I] bf16 input
constexpr size_t OFF_WHB  = OFF_INBF + (size_t)kT * kB * kI * 2; // [H, H] bf16 (row n, col k)
constexpr size_t OFF_WIB  = OFF_WHB  + (size_t)kH * kH * 2;      // [H, I] bf16
constexpr size_t OFF_WOB  = OFF_WIB  + (size_t)kH * kI * 2;      // [O, H] bf16
constexpr size_t OFF_BSUM = OFF_WOB  + (size_t)kO * kH * 2;      // [H] f32 (bi+bh)
constexpr size_t OFF_RBF  = OFF_BSUM + (size_t)kH * 4;           // 2 x [B, H] bf16 state (double buffer)
constexpr size_t OFF_SYNC = OFF_RBF  + (size_t)2 * kB * kH * 2;  // barrier counters

__device__ __forceinline__ unsigned short f2bf_bits(float f) {
  unsigned u = __float_as_uint(f);
  u += 0x7FFFu + ((u >> 16) & 1u);   // round-to-nearest-even
  return (unsigned short)(u >> 16);
}

__device__ __forceinline__ float fast_tanh(float x) {
  float e = __expf(2.0f * x);                       // v_exp_f32
  return 1.0f - 2.0f * __builtin_amdgcn_rcpf(e + 1.0f);
}

// A-tile (16x32 bf16, M x K): lane L -> row m0+(L&15); K chunks {kb..kb+7, kb+16..kb+23}, kb = (L>=16)?8:0
__device__ __forceinline__ v16bf load_tileA(const unsigned short* __restrict__ base,
                                            int stride, int m0, int k0, int lane) {
  int m  = m0 + (lane & 15);
  int kb = k0 + ((lane >> 4) << 3);
  const unsigned short* p = base + (size_t)m * stride + kb;
  v8bf a0 = *(const v8bf*)(p);
  v8bf a1 = *(const v8bf*)(p + 16);
  v16bf a;
#pragma unroll
  for (int i = 0; i < 8; ++i) { a[i] = a0[i]; a[8 + i] = a1[i]; }
  return a;
}

// B-tile (32x16 bf16, K x N): lane L -> col n0+(L&15); K chunks {kb..kb+7, kb+16..kb+23}, kb = (L>=16)?8:0
// Memory layout: Bmat[n][k] row-major -> two 16-byte chunks per lane.
__device__ __forceinline__ v16bf load_tileB(const unsigned short* __restrict__ base,
                                            int stride, int n0, int k0, int lane) {
  int n  = n0 + (lane & 15);
  int kb = k0 + ((lane >> 4) << 3);
  const unsigned short* p = base + (size_t)n * stride + kb;
  v8bf b0 = *(const v8bf*)(p);
  v8bf b1 = *(const v8bf*)(p + 16);
  v16bf b;
#pragma unroll
  for (int i = 0; i < 8; ++i) { b[i] = b0[i]; b[8 + i] = b1[i]; }
  return b;
}

__device__ __forceinline__ v8f wmma_bf16(v16bf a, v16bf b, v8f c) {
  return __builtin_amdgcn_wmma_f32_16x16x32_bf16(false, a, false, b, (short)0, c, false, false);
}

__device__ __forceinline__ void grid_sync(unsigned* cnt, unsigned* gen, unsigned nblocks) {
  __syncthreads();
  if (threadIdx.x == 0) {
    __threadfence();
    unsigned g = __hip_atomic_load(gen, __ATOMIC_RELAXED, __HIP_MEMORY_SCOPE_AGENT);
    unsigned prev = __hip_atomic_fetch_add(cnt, 1u, __ATOMIC_ACQ_REL, __HIP_MEMORY_SCOPE_AGENT);
    if (prev == nblocks - 1u) {
      __hip_atomic_store(cnt, 0u, __ATOMIC_RELAXED, __HIP_MEMORY_SCOPE_AGENT);
      __hip_atomic_fetch_add(gen, 1u, __ATOMIC_ACQ_REL, __HIP_MEMORY_SCOPE_AGENT);
    } else {
      while (__hip_atomic_load(gen, __ATOMIC_ACQUIRE, __HIP_MEMORY_SCOPE_AGENT) == g) {
        __builtin_amdgcn_s_sleep(1);
      }
    }
  }
  __syncthreads();
}

// ---- phase 0: convert weights/inputs to bf16, build bias sum, init state buffer 0 ----
__global__ void convert_all(const float* __restrict__ in, const float* __restrict__ rate,
                            const float* __restrict__ Wi, const float* __restrict__ bi,
                            const float* __restrict__ Wh, const float* __restrict__ bh,
                            const float* __restrict__ Wo,
                            unsigned short* __restrict__ inbf, unsigned short* __restrict__ Whb,
                            unsigned short* __restrict__ Wib, unsigned short* __restrict__ Wob,
                            float* __restrict__ bsum, unsigned short* __restrict__ rbf0) {
  size_t i0 = (size_t)blockIdx.x * blockDim.x + threadIdx.x;
  size_t stride = (size_t)gridDim.x * blockDim.x;
  for (size_t k = i0; k < (size_t)kT * kB * kI; k += stride) inbf[k] = f2bf_bits(in[k]);
  for (size_t k = i0; k < (size_t)kH * kH;      k += stride) Whb[k]  = f2bf_bits(Wh[k]);
  for (size_t k = i0; k < (size_t)kH * kI;      k += stride) Wib[k]  = f2bf_bits(Wi[k]);
  for (size_t k = i0; k < (size_t)kO * kH;      k += stride) Wob[k]  = f2bf_bits(Wo[k]);
  for (size_t k = i0; k < (size_t)kH;           k += stride) bsum[k] = bi[k] + bh[k];
  for (size_t k = i0; k < (size_t)kB * kH;      k += stride) rbf0[k] = f2bf_bits(rate[k]);
}

// ---- phase B: persistent recurrent kernel. 64 blocks x 8 waves = 512 waves = 512 C-tiles ----
__global__ __launch_bounds__(256) void rnn_recurrent(
    const unsigned short* __restrict__ inbf, const unsigned short* __restrict__ Whb,
    const unsigned short* __restrict__ Wib, const float* __restrict__ bsum,
    const float* __restrict__ rate0, unsigned short* __restrict__ rbf,
    float* __restrict__ rate_out, unsigned* __restrict__ sync) {
  const int lane = threadIdx.x & 31;
  const int gw   = blockIdx.x * (blockDim.x >> 5) + (threadIdx.x >> 5); // 0..511
  const int m0   = (gw & 3) * 16;          // batch tile
  const int n0   = (gw >> 2) * 16;         // hidden tile
  const int n    = n0 + (lane & 15);
  const int mb   = m0 + ((lane >> 4) << 3);
  const float bs = bsum[n];

  float rstate[8];                         // leaky-integrator carry lives in registers
#pragma unroll
  for (int j = 0; j < 8; ++j) rstate[j] = rate0[(size_t)(mb + j) * kH + n];

  unsigned* cnt = sync;
  unsigned* gen = sync + 16;

  for (int t = 0; t < kT; ++t) {
    const unsigned short* rcur = rbf + (size_t)(t & 1) * (kB * kH);
    unsigned short*       rnxt = rbf + (size_t)((t + 1) & 1) * (kB * kH);

    // two independent accumulators break the WMMA C->D dependency chain
    v8f acc0 = {};
    v8f acc1 = {};
    // recurrent term: r @ Wh.T  (K = H, Wh/r bf16 resident in L2)
    for (int k0 = 0; k0 < kH; k0 += 64) {
      acc0 = wmma_bf16(load_tileA(rcur, kH, m0, k0, lane),
                       load_tileB(Whb, kH, n0, k0, lane), acc0);
      acc1 = wmma_bf16(load_tileA(rcur, kH, m0, k0 + 32, lane),
                       load_tileB(Whb, kH, n0, k0 + 32, lane), acc1);
    }
    // fused input projection: input[t] @ Wi.T  (K = I)
    const unsigned short* irow = inbf + (size_t)t * kB * kI;
    for (int k0 = 0; k0 < kI; k0 += 64) {
      acc0 = wmma_bf16(load_tileA(irow, kI, m0, k0, lane),
                       load_tileB(Wib, kI, n0, k0, lane), acc0);
      acc1 = wmma_bf16(load_tileA(irow, kI, m0, k0 + 32, lane),
                       load_tileB(Wib, kI, n0, k0 + 32, lane), acc1);
    }
    v8f acc = acc0 + acc1;

    float* rout = rate_out + (size_t)t * kB * kH;
#pragma unroll
    for (int j = 0; j < 8; ++j) {
      float rn = 0.9f * rstate[j] + 0.1f * fast_tanh(acc[j] + bs);
      rstate[j] = rn;
      size_t idx = (size_t)(mb + j) * kH + n;
      rnxt[idx] = f2bf_bits(rn);
      rout[idx] = rn;
    }
    grid_sync(cnt, gen, gridDim.x);
  }
}

// ---- phase C: output projection, f32 rate converted to bf16 in-register, WMMA over K=H ----
__global__ __launch_bounds__(256) void out_proj(const float* __restrict__ rate_all,
                                                const unsigned short* __restrict__ Wob,
                                                const float* __restrict__ bo,
                                                float* __restrict__ out) {
  const int lane = threadIdx.x & 31;
  const int gw   = blockIdx.x * (blockDim.x >> 5) + (threadIdx.x >> 5); // 0..2047
  const int m0   = gw * 16;

  v8f acc[4] = {};
  for (int k0 = 0; k0 < kH; k0 += 32) {
    int m  = m0 + (lane & 15);
    int kb = k0 + ((lane >> 4) << 3);
    const float* ap = rate_all + (size_t)m * kH + kb;
    v8f f0 = *(const v8f*)(ap);
    v8f f1 = *(const v8f*)(ap + 16);
    v16bf a;
#pragma unroll
    for (int i = 0; i < 8; ++i) {
      a[i]     = __builtin_bit_cast(__bf16, f2bf_bits(f0[i]));
      a[8 + i] = __builtin_bit_cast(__bf16, f2bf_bits(f1[i]));
    }
#pragma unroll
    for (int nt = 0; nt < 4; ++nt)
      acc[nt] = wmma_bf16(a, load_tileB(Wob, kH, nt * 16, k0, lane), acc[nt]);
  }

  const int nl = lane & 15;
  const int mb = m0 + ((lane >> 4) << 3);
#pragma unroll
  for (int nt = 0; nt < 4; ++nt) {
    float bov = bo[nt * 16 + nl];
#pragma unroll
    for (int j = 0; j < 8; ++j)
      out[(size_t)(mb + j) * kO + nt * 16 + nl] = acc[nt][j] + bov;
  }
}

extern "C" void kernel_launch(void* const* d_in, const int* in_sizes, int n_in,
                              void* d_out, int out_size, void* d_ws, size_t ws_size,
                              hipStream_t stream) {
  (void)in_sizes; (void)n_in; (void)out_size; (void)ws_size;
  const float* input_sig = (const float*)d_in[0];
  const float* rate      = (const float*)d_in[1];
  const float* Wi        = (const float*)d_in[2];
  const float* bi        = (const float*)d_in[3];
  const float* Wh        = (const float*)d_in[4];
  const float* bh        = (const float*)d_in[5];
  const float* Wo        = (const float*)d_in[6];
  const float* bo        = (const float*)d_in[7];

  char* ws = (char*)d_ws;
  unsigned short* inbf = (unsigned short*)(ws + OFF_INBF);
  unsigned short* Whb  = (unsigned short*)(ws + OFF_WHB);
  unsigned short* Wib  = (unsigned short*)(ws + OFF_WIB);
  unsigned short* Wob  = (unsigned short*)(ws + OFF_WOB);
  float*          bsum = (float*)(ws + OFF_BSUM);
  unsigned short* rbf  = (unsigned short*)(ws + OFF_RBF);
  unsigned*       sync = (unsigned*)(ws + OFF_SYNC);

  float* out      = (float*)d_out;                         // [T,B,O]
  float* rate_all = out + (size_t)kT * kB * kO;            // [T,B,H]

  hipMemsetAsync(ws + OFF_SYNC, 0, 256, stream);           // reset barrier state every call

  convert_all<<<2048, 256, 0, stream>>>(input_sig, rate, Wi, bi, Wh, bh, Wo,
                                        inbf, Whb, Wib, Wob, bsum, rbf);
  rnn_recurrent<<<64, 256, 0, stream>>>(inbf, Whb, Wib, bsum, rate, rbf, rate_all, sync);
  out_proj<<<256, 256, 0, stream>>>(rate_all, Wob, bo, out);
}
